// MolecularGNN_45595372814708
// MI455X (gfx1250) — matrix operand order repaired and checked by reading, not compiled, not run on hardware
//
#include <hip/hip_runtime.h>
#include <hip/hip_bf16.h>
#include <math.h>

typedef __bf16 bf16;
typedef __attribute__((ext_vector_type(16))) __bf16 v16bf;
typedef __attribute__((ext_vector_type(8)))  __bf16 v8bf;
typedef __attribute__((ext_vector_type(4)))  __bf16 v4bf;
typedef __attribute__((ext_vector_type(8)))  float  v8f;
typedef __attribute__((ext_vector_type(4)))  float  v4f;

#define LDA 40   // bf16 elems per LDS row (A tile); 80B rows keep 16B-aligned sublanes
#define LDB 40   // bf16 elems per LDS row (B tile, stored transposed [n][k])

// ---------------- utility kernels ----------------

__global__ void fill_f32(float* __restrict__ p, float v, long long n) {
    long long i = (long long)blockIdx.x * blockDim.x + threadIdx.x;
    if (i < n) p[i] = v;
}

__global__ void deg_accum(const int* __restrict__ dst, float* __restrict__ deg, int E) {
    int e = blockIdx.x * blockDim.x + threadIdx.x;
    if (e < E) unsafeAtomicAdd(&deg[dst[e]], 1.0f);
}

__global__ void to_rsqrt(float* __restrict__ p, int n) {
    int i = blockIdx.x * blockDim.x + threadIdx.x;
    if (i < n) p[i] = rsqrtf(p[i]);
}

// p[n*C + c] = b[c]; C is a power of two (cmask = C-1)
__global__ void init_bias(float* __restrict__ p, const float* __restrict__ b,
                          long long total, int cmask) {
    long long i = (long long)blockIdx.x * blockDim.x + threadIdx.x;
    if (i < total) p[i] = b[(int)i & cmask];
}

// ---------------- bf16 WMMA GEMM with fused BN+ReLU on A ----------------
// C[M,NC] = act(BN(A))[M,K] @ B[K,NC]   (no bias; scl==nullptr -> raw A)
__global__ __launch_bounds__(256)
void gemm_bf16_wmma(const float* __restrict__ A, const float* __restrict__ B,
                    float* __restrict__ C, int M, int K, int NC,
                    const float* __restrict__ scl, const float* __restrict__ shf,
                    int relu)
{
    __shared__ alignas(16) bf16 lA[128 * LDA];
    __shared__ alignas(16) bf16 lB[64 * LDB];

    const int tid  = threadIdx.x;
    const int lane = tid & 31;
    const int lm   = lane & 15;
    const int hf   = lane >> 4;      // lane half (ISA fragment layout)
    const int wid  = tid >> 5;
    const int wM   = wid & 3;        // 4 wave-rows
    const int wN   = wid >> 2;       // 2 wave-cols

    const int m0 = blockIdx.x * 128;
    const int n0 = blockIdx.y * 64;

    v8f acc[2][2] = {};

    for (int kb = 0; kb < K; kb += 32) {
        __syncthreads();
        // ---- stage A tile (128x32): float4 loads, BN affine, pack -> ds_store_b64
        #pragma unroll
        for (int i = 0; i < 4; ++i) {
            int q  = i * 256 + tid;        // float4 index within tile (1024 total)
            int r  = q >> 3;               // 8 float4 per 32-wide row
            int c4 = (q & 7) * 4;
            int gr = m0 + r;
            v4f v = {0.f, 0.f, 0.f, 0.f};
            if (gr < M) v = *(const v4f*)&A[(size_t)gr * K + kb + c4];
            if (scl) {
                v4f sc = *(const v4f*)&scl[kb + c4];
                v4f sh = *(const v4f*)&shf[kb + c4];
                v = v * sc + sh;
                if (relu) {
                    v.x = fmaxf(v.x, 0.f); v.y = fmaxf(v.y, 0.f);
                    v.z = fmaxf(v.z, 0.f); v.w = fmaxf(v.w, 0.f);
                }
            }
            v4bf pk = { (bf16)v.x, (bf16)v.y, (bf16)v.z, (bf16)v.w };
            *(v4bf*)&lA[r * LDA + c4] = pk;    // 8B-aligned (row stride 80B)
        }
        // ---- stage B tile (32x64) transposed: float4 loads, 4x b16 stores
        #pragma unroll
        for (int i = 0; i < 2; ++i) {
            int q  = i * 256 + tid;        // float4 index (512 total)
            int k  = q >> 4;               // 16 float4 per 64-wide row
            int n4 = (q & 15) * 4;
            v4f v = *(const v4f*)&B[(size_t)(kb + k) * NC + n0 + n4];
            lB[(n4 + 0) * LDB + k] = (bf16)v.x;
            lB[(n4 + 1) * LDB + k] = (bf16)v.y;
            lB[(n4 + 2) * LDB + k] = (bf16)v.z;
            lB[(n4 + 3) * LDB + k] = (bf16)v.w;
        }
        __syncthreads();

        // A fragment: lane half h holds K = {8h..8h+7, 16+8h..16+8h+7}
        v16bf afr[2], bfr[2];
        #pragma unroll
        for (int mi = 0; mi < 2; ++mi) {
            int r = wM * 32 + mi * 16 + lm;
            v8bf a0 = *(const v8bf*)&lA[r * LDA + 8 * hf];
            v8bf a1 = *(const v8bf*)&lA[r * LDA + 16 + 8 * hf];
            afr[mi] = __builtin_shufflevector(a0, a1, 0,1,2,3,4,5,6,7,8,9,10,11,12,13,14,15);
        }
        // B fragment: lane half h holds K = {8h..8h+7, 16+8h..16+8h+7}
        #pragma unroll
        for (int ni = 0; ni < 2; ++ni) {
            int c = wN * 32 + ni * 16 + lm;
            v8bf b0 = *(const v8bf*)&lB[c * LDB + 8 * hf];
            v8bf b1 = *(const v8bf*)&lB[c * LDB + 16 + 8 * hf];
            bfr[ni] = __builtin_shufflevector(b0, b1, 0,1,2,3,4,5,6,7,8,9,10,11,12,13,14,15);
        }
        #pragma unroll
        for (int mi = 0; mi < 2; ++mi)
            #pragma unroll
            for (int ni = 0; ni < 2; ++ni)
                acc[mi][ni] = __builtin_amdgcn_wmma_f32_16x16x32_bf16(
                    false, afr[mi], false, bfr[ni], (short)0, acc[mi][ni], false, false);
    }

    // D layout: VGPR r -> M = r + 8*half, N = lane&15
    #pragma unroll
    for (int mi = 0; mi < 2; ++mi) {
        int rbase = m0 + wM * 32 + mi * 16 + hf * 8;
        #pragma unroll
        for (int ni = 0; ni < 2; ++ni) {
            int col = n0 + wN * 32 + ni * 16 + lm;
            #pragma unroll
            for (int r = 0; r < 8; ++r) {
                int row = rbase + r;
                if (row < M) C[(size_t)row * NC + col] = acc[mi][ni][r];
            }
        }
    }
}

// ---------------- edge aggregation: out[dst] += norm * h[src] ----------------
// items [0,E) are edges; items [E, E+N) are self-loops (norm = dis^2)
__global__ __launch_bounds__(256)
void agg_edges(const float* __restrict__ h, const int* __restrict__ src,
               const int* __restrict__ dst, const float* __restrict__ dis,
               float* __restrict__ out, int E, int N, int C)
{
    long long item = (long long)blockIdx.x * 8 + (threadIdx.x >> 5);
    int lane = threadIdx.x & 31;
    long long T = (long long)E + N;
    if (item >= T) return;
    int s, d; float nrm;
    if (item < E) {
        s = src[item]; d = dst[item];
        nrm = dis[s] * dis[d];
    } else {
        s = d = (int)(item - E);
        float x = dis[s];
        nrm = x * x;
    }
    const float* hs = h + (size_t)s * C;
    float* od = out + (size_t)d * C;
    for (int c = lane; c < C; c += 32)
        unsafeAtomicAdd(&od[c], nrm * hs[c]);
}

// ---------------- BatchNorm statistics ----------------
__global__ void bn_stats(const float* __restrict__ h, float* __restrict__ sums,
                         float* __restrict__ sumsq, int Nn, int C)
{
    int c  = threadIdx.x;            // blockDim.x == C
    int r0 = blockIdx.x * 64;
    float s = 0.f, ss = 0.f;
    for (int r = 0; r < 64; ++r) {
        int row = r0 + r;
        if (row < Nn) {
            float v = h[(size_t)row * C + c];
            s += v; ss += v * v;
        }
    }
    unsafeAtomicAdd(&sums[c], s);
    unsafeAtomicAdd(&sumsq[c], ss);
}

__global__ void bn_final(const float* __restrict__ sums, const float* __restrict__ sumsq,
                         const float* __restrict__ g, const float* __restrict__ be,
                         float* __restrict__ scl, float* __restrict__ shf,
                         float ninv, int C)
{
    int c = blockIdx.x * blockDim.x + threadIdx.x;
    if (c < C) {
        float mu  = sums[c] * ninv;
        float var = fmaxf(sumsq[c] * ninv - mu * mu, 0.0f);
        float sc  = g[c] * rsqrtf(var + 1e-5f);
        scl[c] = sc;
        shf[c] = be[c] - mu * sc;
    }
}

// ---------------- global mean pool (BN of layer3 fused) ----------------
__global__ __launch_bounds__(256)
void pool_accum(const float* __restrict__ h, const int* __restrict__ batch,
                const float* __restrict__ scl, const float* __restrict__ shf,
                float* __restrict__ pool, float* __restrict__ cnt, int Nn, int C)
{
    int node = blockIdx.x * 8 + (threadIdx.x >> 5);
    int lane = threadIdx.x & 31;
    if (node >= Nn) return;
    int g = batch[node];
    const float* hp = h + (size_t)node * C;
    float* pp = pool + (size_t)g * C;
    for (int c = lane; c < C; c += 32)
        unsafeAtomicAdd(&pp[c], hp[c] * scl[c] + shf[c]);
    if (lane == 0) unsafeAtomicAdd(&cnt[g], 1.0f);
}

__global__ void l2_finalize(const float* __restrict__ pool, const float* __restrict__ cnt,
                            float* __restrict__ out, int D)
{
    __shared__ float red[256];
    int g = blockIdx.x;
    int c = threadIdx.x;             // blockDim.x == D
    float v = pool[(size_t)g * D + c] / fmaxf(cnt[g], 1.0f);
    red[c] = v * v;
    __syncthreads();
    for (int s = D >> 1; s > 0; s >>= 1) {
        if (c < s) red[c] += red[c + s];
        __syncthreads();
    }
    float nrm = fmaxf(sqrtf(red[0]), 1e-12f);
    out[(size_t)g * D + c] = v / nrm;
}

// ---------------- orchestration ----------------
extern "C" void kernel_launch(void* const* d_in, const int* in_sizes, int n_in,
                              void* d_out, int out_size, void* d_ws, size_t ws_size,
                              hipStream_t stream)
{
    const float* x   = (const float*)d_in[0];
    const int*   ei  = (const int*)d_in[1];
    const int*   bat = (const int*)d_in[2];
    const float* W1  = (const float*)d_in[3];
    const float* b1  = (const float*)d_in[4];
    const float* g1  = (const float*)d_in[5];
    const float* be1 = (const float*)d_in[6];
    const float* W2  = (const float*)d_in[7];
    const float* b2  = (const float*)d_in[8];
    const float* g2  = (const float*)d_in[9];
    const float* be2 = (const float*)d_in[10];
    const float* W3  = (const float*)d_in[11];
    const float* b3  = (const float*)d_in[12];
    const float* g3  = (const float*)d_in[13];
    const float* be3 = (const float*)d_in[14];
    float* out = (float*)d_out;

    const int N = in_sizes[2];
    const int F = in_sizes[0] / N;
    const int E = in_sizes[1] / 2;
    const int H = in_sizes[4];
    const int D = in_sizes[12];
    const int G = out_size / D;

    const int* src = ei;
    const int* dst = ei + E;

    char* w = (char*)d_ws;
    auto carve = [&](size_t bytes) -> float* {
        float* p = (float*)w;
        w += (bytes + 255) & ~(size_t)255;
        return p;
    };
    float* dis   = carve((size_t)N * 4);
    float* buf0  = carve((size_t)N * H * 4);
    float* buf1  = carve((size_t)N * H * 4);
    float* sums  = carve((size_t)H * 4);
    float* sumsq = carve((size_t)H * 4);
    float* scl   = carve((size_t)H * 4);
    float* shf   = carve((size_t)H * 4);
    float* pool  = carve((size_t)G * D * 4);
    float* cnt   = carve((size_t)G * 4);

    dim3 blk(256);
    auto cdiv = [](long long a, long long b) { return (int)((a + b - 1) / b); };
    const float ninv = 1.0f / (float)N;
    const long long T = (long long)E + N;

    // GCN norm: deg (incl. self-loop) -> dis = rsqrt(deg)
    fill_f32 <<<cdiv(N, 256), blk, 0, stream>>>(dis, 1.0f, N);
    deg_accum<<<cdiv(E, 256), blk, 0, stream>>>(dst, dis, E);
    to_rsqrt <<<cdiv(N, 256), blk, 0, stream>>>(dis, N);

    // -------- layer 1: conv -> BN stats --------
    gemm_bf16_wmma<<<dim3(cdiv(N,128), H/64), blk, 0, stream>>>(x, W1, buf0, N, F, H, nullptr, nullptr, 0);
    init_bias<<<cdiv((long long)N*H,256), blk, 0, stream>>>(buf1, b1, (long long)N*H, H - 1);
    agg_edges<<<cdiv(T,8), blk, 0, stream>>>(buf0, src, dst, dis, buf1, E, N, H);
    fill_f32<<<cdiv(H,256), blk, 0, stream>>>(sums, 0.f, H);
    fill_f32<<<cdiv(H,256), blk, 0, stream>>>(sumsq, 0.f, H);
    bn_stats<<<cdiv(N,64), dim3(H), 0, stream>>>(buf1, sums, sumsq, N, H);
    bn_final<<<1, dim3(H), 0, stream>>>(sums, sumsq, g1, be1, scl, shf, ninv, H);

    // -------- layer 2 (BN1+ReLU fused into GEMM A-load) --------
    gemm_bf16_wmma<<<dim3(cdiv(N,128), H/64), blk, 0, stream>>>(buf1, W2, buf0, N, H, H, scl, shf, 1);
    init_bias<<<cdiv((long long)N*H,256), blk, 0, stream>>>(buf1, b2, (long long)N*H, H - 1);
    agg_edges<<<cdiv(T,8), blk, 0, stream>>>(buf0, src, dst, dis, buf1, E, N, H);
    fill_f32<<<cdiv(H,256), blk, 0, stream>>>(sums, 0.f, H);
    fill_f32<<<cdiv(H,256), blk, 0, stream>>>(sumsq, 0.f, H);
    bn_stats<<<cdiv(N,64), dim3(H), 0, stream>>>(buf1, sums, sumsq, N, H);
    bn_final<<<1, dim3(H), 0, stream>>>(sums, sumsq, g2, be2, scl, shf, ninv, H);

    // -------- layer 3 (BN2+ReLU fused; no ReLU after) --------
    gemm_bf16_wmma<<<dim3(cdiv(N,128), D/64), blk, 0, stream>>>(buf1, W3, buf0, N, H, D, scl, shf, 1);
    init_bias<<<cdiv((long long)N*D,256), blk, 0, stream>>>(buf1, b3, (long long)N*D, D - 1);
    agg_edges<<<cdiv(T,8), blk, 0, stream>>>(buf0, src, dst, dis, buf1, E, N, D);
    fill_f32<<<cdiv(D,256), blk, 0, stream>>>(sums, 0.f, D);
    fill_f32<<<cdiv(D,256), blk, 0, stream>>>(sumsq, 0.f, D);
    bn_stats<<<cdiv(N,64), dim3(D), 0, stream>>>(buf1, sums, sumsq, N, D);
    bn_final<<<1, dim3(D), 0, stream>>>(sums, sumsq, g3, be3, scl, shf, ninv, D);

    // -------- mean pool (BN3 fused) + L2 normalize --------
    fill_f32<<<cdiv((long long)G*D,256), blk, 0, stream>>>(pool, 0.f, (long long)G*D);
    fill_f32<<<cdiv(G,256), blk, 0, stream>>>(cnt, 0.f, G);
    pool_accum<<<cdiv(N,8), blk, 0, stream>>>(buf1, bat, scl, shf, pool, cnt, N, D);
    l2_finalize<<<G, dim3(D), 0, stream>>>(pool, cnt, out, D);
}
